// RBFLayer_13932873909049
// MI455X (gfx1250) — compile-verified, matrix operand order checked by reading, not run on hardware
//
#include <hip/hip_runtime.h>
#include <hip/hip_bf16.h>
#include <stdint.h>

// Problem dimensions are fixed by the reference.
#define B_DIM 16384
#define I_DIM 1024
#define O_DIM 4096

// GEMM tiling.
#define BM 128              // workgroup tile M
#define BN 128              // workgroup tile N
#define KT 64               // K elements staged in LDS per buffer
#define PITCH 72            // LDS row pitch in elements (64 + 8 pad -> conflict-free)
#define TILE_ELEMS (128 * PITCH)              // one sub-tile (A-hi etc.)
#define TILE_BYTES (TILE_ELEMS * 2)           // 18432 B
#define BUF_ELEMS  (4 * TILE_ELEMS)           // Ahi, Alo, Bhi, Blo
#define BUF_BYTES  (4 * TILE_BYTES)           // 73728 B
#define LDS_TOTAL  (2 * BUF_BYTES)            // double buffered: 147456 B

typedef __attribute__((ext_vector_type(16))) __bf16    v16bf;
typedef __attribute__((ext_vector_type(8)))  float     v8f;
typedef __attribute__((ext_vector_type(4)))  uint32_t  v4u;

static __device__ __forceinline__ uint16_t f32_to_bf16(float f) {
    uint32_t u = __float_as_uint(f);
    u += 0x7FFFu + ((u >> 16) & 1u);   // round-to-nearest-even
    return (uint16_t)(u >> 16);
}
static __device__ __forceinline__ float bf16_to_f32(uint16_t h) {
    return __uint_as_float(((uint32_t)h) << 16);
}

// ---------------------------------------------------------------------------
// CDNA5 async copy: global -> LDS, 16 bytes per lane, tracked by ASYNCcnt.
// vdst = per-lane LDS byte address (low 32 bits of flat shared address),
// vaddr = per-lane 64-bit global address.
// ---------------------------------------------------------------------------
static __device__ __forceinline__ void async_copy_b128(unsigned lds_byte, uint64_t gaddr) {
    asm volatile("global_load_async_to_lds_b128 %0, %1, off"
                 :: "v"(lds_byte), "v"(gaddr) : "memory");
}
static __device__ __forceinline__ void wait_asynccnt0() {
    asm volatile("s_wait_asynccnt 0" ::: "memory");
}

// ---------------------------------------------------------------------------
// Prep: per row, compute sum of squares, split f32 -> bf16 hi/lo pair.
// One wave per row; 8 waves (256 threads) per block.
// For centers also precompute inv = 1/(2*sigma^2) = 0.5*exp(-2*log_sigma).
// ---------------------------------------------------------------------------
__global__ void rbf_prep_rows(const float* __restrict__ src, int rows,
                              uint16_t* __restrict__ hi, uint16_t* __restrict__ lo,
                              float* __restrict__ sq,
                              const float* __restrict__ log_sigma,
                              float* __restrict__ inv2s) {
    const int lane = threadIdx.x & 31;
    const int wave = threadIdx.x >> 5;
    const int row  = blockIdx.x * 8 + wave;
    if (row >= rows) return;

    const float4* srow = (const float4*)(src + (size_t)row * I_DIM);
    uint16_t* hrow = hi + (size_t)row * I_DIM;
    uint16_t* lrow = lo + (size_t)row * I_DIM;

    float s = 0.f;
#pragma unroll
    for (int j = 0; j < (I_DIM / 4) / 32; ++j) {
        const int i4 = lane + j * 32;
        float4 f = srow[i4];
        s += f.x * f.x + f.y * f.y + f.z * f.z + f.w * f.w;

        uint16_t h0 = f32_to_bf16(f.x), h1 = f32_to_bf16(f.y);
        uint16_t h2 = f32_to_bf16(f.z), h3 = f32_to_bf16(f.w);
        uint16_t l0 = f32_to_bf16(f.x - bf16_to_f32(h0));
        uint16_t l1 = f32_to_bf16(f.y - bf16_to_f32(h1));
        uint16_t l2 = f32_to_bf16(f.z - bf16_to_f32(h2));
        uint16_t l3 = f32_to_bf16(f.w - bf16_to_f32(h3));

        ushort4 hv; hv.x = h0; hv.y = h1; hv.z = h2; hv.w = h3;
        ushort4 lv; lv.x = l0; lv.y = l1; lv.z = l2; lv.w = l3;
        *(ushort4*)(hrow + (size_t)i4 * 4) = hv;
        *(ushort4*)(lrow + (size_t)i4 * 4) = lv;
    }
#pragma unroll
    for (int off = 16; off; off >>= 1) s += __shfl_xor(s, off, 32);

    if (lane == 0) {
        sq[row] = s;
        if (log_sigma) inv2s[row] = 0.5f * __expf(-2.0f * log_sigma[row]);
    }
}

// ---------------------------------------------------------------------------
// Fragment load from LDS matching the CDNA5 16-bit A/B WMMA layout:
// lanes 0-15  hold K = [k..k+7] (v0-3) and [k+16..k+23] (v4-7)
// lanes 16-31 hold K = [k+8..k+15]     and [k+24..k+31]
// Caller passes p already offset by kSel = (lane>>4)*8 elements.
// ---------------------------------------------------------------------------
static __device__ __forceinline__ v16bf frag_lds(const uint16_t* p) {
    union { v4u u[2]; v16bf b; } t;
    t.u[0] = *(const v4u*)(p);         // K sub-chunk [0..7]
    t.u[1] = *(const v4u*)(p + 16);    // K sub-chunk [16..23]
    return t.b;
}

// Issue the async fill of one LDS buffer (A-hi/A-lo/B-hi/B-lo tiles for one
// 128xKT / KTx128 K-slab). 16 async b128 instructions per thread.
static __device__ __forceinline__ void issue_fill(
        const uint16_t* __restrict__ aHiB, const uint16_t* __restrict__ aLoB,
        const uint16_t* __restrict__ bHiB, const uint16_t* __restrict__ bLoB,
        int k0, unsigned ldsBufByte, int tid) {
    const uint16_t* srcs[4] = { aHiB, aLoB, bHiB, bLoB };
#pragma unroll
    for (int t = 0; t < 4; ++t) {
        const uint16_t* base = srcs[t];
        const unsigned ldsTile = ldsBufByte + t * TILE_BYTES;
#pragma unroll
        for (int c = 0; c < 4; ++c) {
            const int ci  = c * 256 + tid;      // 1024 chunks of 8 elems per tile
            const int row = ci >> 3;
            const int sub = ci & 7;
            const uint64_t g = (uint64_t)(base + (size_t)row * I_DIM + k0 + sub * 8);
            const unsigned l = ldsTile + row * (PITCH * 2) + sub * 16;
            async_copy_b128(l, g);
        }
    }
}

// ---------------------------------------------------------------------------
// Fused RBF GEMM: cross = x . centers^T via bf16 hi/lo split (3 WMMA terms),
// LDS double-buffered with GLOBAL_LOAD_ASYNC_TO_LDS_B128 staging.
// epilogue: out = exp(-max(xsq + csq - 2*cross, 0) * inv2s)
// Block: 256 threads = 8 waves, WG tile 128x128, wave tile 32x64.
// ---------------------------------------------------------------------------
__global__ void __launch_bounds__(256)
rbf_wmma_kernel(const uint16_t* __restrict__ xHi, const uint16_t* __restrict__ xLo,
                const uint16_t* __restrict__ cHi, const uint16_t* __restrict__ cLo,
                const float* __restrict__ xSq, const float* __restrict__ cSq,
                const float* __restrict__ inv2s, float* __restrict__ out) {
    extern __shared__ char smem_raw[];
    uint16_t* smem = (uint16_t*)smem_raw;
    // Low 32 bits of a flat shared-memory address ARE the LDS byte address
    // (ISA 10.2).
    const unsigned ldsBase = (unsigned)(uintptr_t)smem_raw;

    const int tid   = threadIdx.x;
    const int lane  = tid & 31;
    const int wave  = tid >> 5;
    const int mQuad = wave >> 1;            // 0..3 -> M offset inside WG tile
    const int nHalf = wave & 1;             // 0..1 -> N offset inside WG tile

    const int blkM = blockIdx.y * BM;
    const int blkN = blockIdx.x * BN;

    const int fr   = lane & 15;             // row/col within a 16-wide fragment
    const int kSel = (lane >> 4) * 8;       // per-half-wave K sub-offset

    const uint16_t* aHiB = xHi + (size_t)blkM * I_DIM;
    const uint16_t* aLoB = xLo + (size_t)blkM * I_DIM;
    const uint16_t* bHiB = cHi + (size_t)blkN * I_DIM;
    const uint16_t* bLoB = cLo + (size_t)blkN * I_DIM;

    v8f zero = {};
    v8f acc[2][4];
#pragma unroll
    for (int mt = 0; mt < 2; ++mt)
#pragma unroll
        for (int nt = 0; nt < 4; ++nt) acc[mt][nt] = zero;

    // Prologue: start filling buffer 0 with the first K-slab.
    issue_fill(aHiB, aLoB, bHiB, bLoB, 0, ldsBase, tid);

    const int NIT = I_DIM / KT;             // 16 K-slabs
    for (int kt = 0; kt < NIT; ++kt) {
        const int cur = kt & 1;

        wait_asynccnt0();                   // my copies of cur slab done
        __syncthreads();                    // everyone's copies done; prev reads done

        if (kt + 1 < NIT)                   // kick off next slab into other buffer
            issue_fill(aHiB, aLoB, bHiB, bLoB, (kt + 1) * KT,
                       ldsBase + (1 - cur) * BUF_BYTES, tid);

        const uint16_t* tA_hi = smem + (size_t)cur * BUF_ELEMS;
        const uint16_t* tA_lo = tA_hi + TILE_ELEMS;
        const uint16_t* tB_hi = tA_lo + TILE_ELEMS;
        const uint16_t* tB_lo = tB_hi + TILE_ELEMS;

#pragma unroll
        for (int s = 0; s < 2; ++s) {       // two K=32 WMMA sub-steps per slab
            const int ko = s * 32 + kSel;
            v16bf aHi[2], aLo[2], bHi[4], bLo[4];
#pragma unroll
            for (int mt = 0; mt < 2; ++mt) {
                const int r = mQuad * 32 + mt * 16 + fr;
                aHi[mt] = frag_lds(tA_hi + r * PITCH + ko);
                aLo[mt] = frag_lds(tA_lo + r * PITCH + ko);
            }
#pragma unroll
            for (int nt = 0; nt < 4; ++nt) {
                const int r = nHalf * 64 + nt * 16 + fr;
                bHi[nt] = frag_lds(tB_hi + r * PITCH + ko);
                bLo[nt] = frag_lds(tB_lo + r * PITCH + ko);
            }
#pragma unroll
            for (int mt = 0; mt < 2; ++mt) {
#pragma unroll
                for (int nt = 0; nt < 4; ++nt) {
                    acc[mt][nt] = __builtin_amdgcn_wmma_f32_16x16x32_bf16(
                        false, aHi[mt], false, bHi[nt], (short)0, acc[mt][nt], false, false);
                    acc[mt][nt] = __builtin_amdgcn_wmma_f32_16x16x32_bf16(
                        false, aHi[mt], false, bLo[nt], (short)0, acc[mt][nt], false, false);
                    acc[mt][nt] = __builtin_amdgcn_wmma_f32_16x16x32_bf16(
                        false, aLo[mt], false, bHi[nt], (short)0, acc[mt][nt], false, false);
                }
            }
        }
    }

    // Epilogue. C/D layout: lanes 0-15 -> N=lane, M=vgpr; lanes 16-31 -> M=8+vgpr.
    const int tileM = blkM + mQuad * 32;
    const int tileN = blkN + nHalf * 64;
    const int mSub  = (lane >> 4) * 8;
#pragma unroll
    for (int mt = 0; mt < 2; ++mt) {
#pragma unroll
        for (int nt = 0; nt < 4; ++nt) {
            const int col  = tileN + nt * 16 + fr;
            const float cs = cSq[col];
            const float iv = inv2s[col];
#pragma unroll
            for (int r = 0; r < 8; ++r) {
                const int row = tileM + mt * 16 + mSub + r;
                float d2 = xSq[row] + cs - 2.0f * acc[mt][nt][r];
                d2 = fmaxf(d2, 0.0f);
                out[(size_t)row * O_DIM + col] = __expf(-d2 * iv);
            }
        }
    }
}

// ---------------------------------------------------------------------------
extern "C" void kernel_launch(void* const* d_in, const int* in_sizes, int n_in,
                              void* d_out, int out_size, void* d_ws, size_t ws_size,
                              hipStream_t stream) {
    (void)in_sizes; (void)n_in; (void)out_size; (void)ws_size;

    const float* x   = (const float*)d_in[0];   // [B, I]
    const float* cen = (const float*)d_in[1];   // [O, I]
    const float* ls  = (const float*)d_in[2];   // [O]
    float* out = (float*)d_out;                 // [B, O]

    char* w = (char*)d_ws;
    uint16_t* xHi = (uint16_t*)w;  w += (size_t)B_DIM * I_DIM * sizeof(uint16_t);
    uint16_t* xLo = (uint16_t*)w;  w += (size_t)B_DIM * I_DIM * sizeof(uint16_t);
    uint16_t* cHi = (uint16_t*)w;  w += (size_t)O_DIM * I_DIM * sizeof(uint16_t);
    uint16_t* cLo = (uint16_t*)w;  w += (size_t)O_DIM * I_DIM * sizeof(uint16_t);
    float* xSq   = (float*)w;      w += (size_t)B_DIM * sizeof(float);
    float* cSq   = (float*)w;      w += (size_t)O_DIM * sizeof(float);
    float* inv2s = (float*)w;      w += (size_t)O_DIM * sizeof(float);

    // Pass 1: split/convert + row norms (x), then centers (+ sigma transform).
    rbf_prep_rows<<<B_DIM / 8, 256, 0, stream>>>(x, B_DIM, xHi, xLo, xSq,
                                                 nullptr, nullptr);
    rbf_prep_rows<<<O_DIM / 8, 256, 0, stream>>>(cen, O_DIM, cHi, cLo, cSq,
                                                 ls, inv2s);

    // Pass 2: fused WMMA GEMM + RBF epilogue (async-LDS double buffered).
    dim3 grid(O_DIM / 128, B_DIM / 128);
    rbf_wmma_kernel<<<grid, 256, LDS_TOTAL, stream>>>(xHi, xLo, cHi, cLo,
                                                      xSq, cSq, inv2s, out);
}
